// ConvolveTensor3body_19765439496448
// MI455X (gfx1250) — matrix operand order repaired and checked by reading, not compiled, not run on hardware
//
#include <hip/hip_runtime.h>

// CDNA5 / gfx1250 fused equivariant tensor-product kernel, v2.
// it3 ⊗ W4 collapsed into two f16 WMMA GEMMs; A-fragment features generated
// on the fly with packed-f16 VOP3P math from pre-converted f16 LDS tables.

typedef __attribute__((ext_vector_type(16))) _Float16 v16h;
typedef __attribute__((ext_vector_type(8)))  _Float16 v8h;
typedef __attribute__((ext_vector_type(8)))  float    v8f;

#define INV8       0.35355339059327373f   // 1/sqrt(8)
#define INV_SQRT3  0.57735026918962576f   // 1/sqrt(3)
#define C5A        0.38729833462074170f   // sqrt(15)/10
#define C5B        0.25819888974716110f   // sqrt(15)/15
#define SSCALE     0.02209708691207961f   // 1/sqrt(2048)
#define VSCALE     0.01804219591217581f   // 1/sqrt(3072)

#define ROWS_PER_BLOCK 128
#define ROW_STRIDE     212   // floats: 128 f32 state + 160 f16 tables (80 fl) + 4 pad
#define LDS_ROW_BASE   384   // first 384 floats: W1s..W3v
#define SHMEM_FLOATS   (LDS_ROW_BASE + ROWS_PER_BLOCK * ROW_STRIDE)

// Per-row LDS layout (bytes from row base):
//   0   S1 f32[8]      32  S2 f32[8]     64  V1 f32[8][3]   160 V2 f32[8][3]
//   256 D12 f32[8][8]
//   512 S3h f16[8]     528 V3h f16[3][8] 576 D13h f16[8][8] 704 D23h f16[8][8]

// ---------------------------------------------------------------------------
// Prep: fold W4s [2048,8] / W4v [3072,8] (f32) into f16 B-fragment layout.
//   lane l: column = l&15, K = (j<8 ? 0 : 16) + (l>>4)*8 + (j&7)  (j = 0..15 within v16h)
// ---------------------------------------------------------------------------
__global__ void fold_w4_kernel(const float* __restrict__ W4s,
                               const float* __restrict__ W4v,
                               _Float16* __restrict__ frag_s,
                               _Float16* __restrict__ frag_v) {
  int t = blockIdx.x * blockDim.x + threadIdx.x;     // 0 .. 5119
  if (t >= (64 + 96) * 32) return;
  int chunk = t >> 5;
  int lane  = t & 31;
  int col   = lane & 15;
  int h     = lane >> 4;
  if (chunk < 64) {
    _Float16* dst = frag_s + (size_t)(chunk * 32 + lane) * 16;
#pragma unroll
    for (int j = 0; j < 16; ++j) {
      int K = (j < 8) ? (h * 8 + j) : (16 + h * 8 + (j - 8));
      float val = (col < 8) ? W4s[(size_t)(chunk * 32 + K) * 8 + col] : 0.0f;
      dst[j] = (_Float16)val;
    }
  } else {
    int kc = chunk - 64;
    _Float16* dst = frag_v + (size_t)(kc * 32 + lane) * 16;
#pragma unroll
    for (int j = 0; j < 16; ++j) {
      int K = (j < 8) ? (h * 8 + j) : (16 + h * 8 + (j - 8));
      float val = (col < 8) ? W4v[(size_t)(kc * 32 + K) * 8 + col] : 0.0f;
      dst[j] = (_Float16)val;
    }
  }
}

__device__ __forceinline__ float silu_act(float z) {
  return z / (1.0f + __expf(-z));
}

// ---------------------------------------------------------------------------
// Main fused kernel: 256 threads (8 wave32), 128 rows per block, 64 blocks.
// ---------------------------------------------------------------------------
__global__ void __launch_bounds__(256)
tp3body_kernel(const float* __restrict__ x,
               const float* __restrict__ W1s, const float* __restrict__ W1v,
               const float* __restrict__ W2s, const float* __restrict__ W2v,
               const float* __restrict__ W3s, const float* __restrict__ W3v,
               const _Float16* __restrict__ frag_s,
               const _Float16* __restrict__ frag_v,
               float* __restrict__ out) {
  extern __shared__ float lds[];
  const int t = threadIdx.x;

  // ---- Phase A: stage the six 8x8 linear weights into LDS -----------------
  for (int i = t; i < 384; i += 256) {
    int g = i >> 6, o = i & 63;
    float w;
    switch (g) {
      case 0:  w = W1s[o]; break;
      case 1:  w = W1v[o]; break;
      case 2:  w = W2s[o]; break;
      case 3:  w = W2v[o]; break;
      case 4:  w = W3s[o]; break;
      default: w = W3v[o]; break;
    }
    lds[i] = w;
  }
  __syncthreads();

  // ---- Phase B: per-row linears + SiLU + tables -> LDS --------------------
  // 2 threads per row (wave-uniform split): bhalf 0 writes f32 prefactor
  // state + f16 S3/V3 tables + D12; bhalf 1 writes f16 D13/D23 tables.
  {
    const int r_local = t & 127;
    const int bhalf   = t >> 7;
    const int rowg    = blockIdx.x * ROWS_PER_BLOCK + r_local;

    float xs[8], xvf[24];
    const float4* xp = (const float4*)(x + (size_t)rowg * 32);
#pragma unroll
    for (int i = 0; i < 2; ++i) {
      float4 q = xp[i];
      xs[i * 4 + 0] = q.x; xs[i * 4 + 1] = q.y;
      xs[i * 4 + 2] = q.z; xs[i * 4 + 3] = q.w;
    }
#pragma unroll
    for (int i = 0; i < 6; ++i) {
      float4 q = xp[2 + i];
      xvf[i * 4 + 0] = q.x; xvf[i * 4 + 1] = q.y;
      xvf[i * 4 + 2] = q.z; xvf[i * 4 + 3] = q.w;
    }

    const float* Lw1s = lds;        const float* Lw1v = lds + 64;
    const float* Lw2s = lds + 128;  const float* Lw2v = lds + 192;
    const float* Lw3s = lds + 256;  const float* Lw3v = lds + 320;

    float n1s[8], n2s[8], n3s[8];
    float n1v[8][3], n2v[8][3], n3v[8][3];
#pragma unroll
    for (int vv = 0; vv < 8; ++vv) {
      float a1 = 0.f, a2 = 0.f, a3 = 0.f;
      float b1[3] = {0.f, 0.f, 0.f}, b2[3] = {0.f, 0.f, 0.f}, b3[3] = {0.f, 0.f, 0.f};
#pragma unroll
      for (int u = 0; u < 8; ++u) {
        float w1 = Lw1s[u * 8 + vv], w2 = Lw2s[u * 8 + vv], w3 = Lw3s[u * 8 + vv];
        a1 += xs[u] * w1; a2 += xs[u] * w2; a3 += xs[u] * w3;
        float q1 = Lw1v[u * 8 + vv], q2 = Lw2v[u * 8 + vv], q3 = Lw3v[u * 8 + vv];
#pragma unroll
        for (int k = 0; k < 3; ++k) {
          float xe = xvf[u * 3 + k];
          b1[k] += xe * q1; b2[k] += xe * q2; b3[k] += xe * q3;
        }
      }
      n1s[vv] = silu_act(a1 * INV8);
      n2s[vv] = silu_act(a2 * INV8);
      n3s[vv] = silu_act(a3 * INV8);
#pragma unroll
      for (int k = 0; k < 3; ++k) {
        n1v[vv][k] = silu_act(b1[k] * INV8);
        n2v[vv][k] = silu_act(b2[k] * INV8);
        n3v[vv][k] = silu_act(b3[k] * INV8);
      }
    }

    float*    R  = lds + LDS_ROW_BASE + r_local * ROW_STRIDE;
    _Float16* Hp = (_Float16*)(R + 128);      // f16 region at +512B
    if (bhalf == 0) {
#pragma unroll
      for (int i = 0; i < 8; ++i) { R[i] = n1s[i]; R[8 + i] = n2s[i]; }
#pragma unroll
      for (int u = 0; u < 8; ++u)
#pragma unroll
        for (int k = 0; k < 3; ++k) {
          R[16 + u * 3 + k] = n1v[u][k];      // V1 at +64B
          R[40 + u * 3 + k] = n2v[u][k];      // V2 at +160B
        }
#pragma unroll
      for (int u = 0; u < 8; ++u)
#pragma unroll
        for (int v = 0; v < 8; ++v)
          R[64 + u * 8 + v] = n1v[u][0] * n2v[v][0] + n1v[u][1] * n2v[v][1] + n1v[u][2] * n2v[v][2];
      // f16 tables: S3h[8], V3h[3][8]
      v8h s3t;
#pragma unroll
      for (int i = 0; i < 8; ++i) s3t[i] = (_Float16)n3s[i];
      *(v8h*)(Hp + 0) = s3t;
#pragma unroll
      for (int k = 0; k < 3; ++k) {
        v8h v3t;
#pragma unroll
        for (int i = 0; i < 8; ++i) v3t[i] = (_Float16)n3v[i][k];
        *(v8h*)(Hp + 8 + 8 * k) = v3t;
      }
    } else {
      // f16 dot tables D13h, D23h (row-contiguous, one b128 store per row)
#pragma unroll
      for (int u = 0; u < 8; ++u) {
        v8h r13, r23;
#pragma unroll
        for (int v3 = 0; v3 < 8; ++v3) {
          float d13 = n1v[u][0] * n3v[v3][0] + n1v[u][1] * n3v[v3][1] + n1v[u][2] * n3v[v3][2];
          float d23 = n2v[u][0] * n3v[v3][0] + n2v[u][1] * n3v[v3][1] + n2v[u][2] * n3v[v3][2];
          r13[v3] = (_Float16)d13;
          r23[v3] = (_Float16)d23;
        }
        *(v8h*)(Hp + 32 + u * 8) = r13;
        *(v8h*)(Hp + 96 + u * 8) = r23;
      }
    }
  }
  __syncthreads();

  // ---- Phase C: WMMA GEMMs with packed-f16 A-fragment generation ----------
  const int lane  = t & 31;
  const int wv    = t >> 5;
  const int m     = lane & 15;          // A-matrix row M handled by this lane
  const int h     = lane >> 4;          // K-half (A layout: h=0 -> K 0..7,16..23)
  const int ncol  = lane & 15;          // D column
  const int obase = blockIdx.x * ROWS_PER_BLOCK + wv * 16;

  const float*    Q    = lds + LDS_ROW_BASE + (wv * 16 + m) * ROW_STRIDE;
  const float*    S1   = Q;        const float* S2 = Q + 8;
  const float*    V1   = Q + 16;   const float* V2 = Q + 40;
  const float*    D12  = Q + 64;
  const _Float16* Hq   = (const _Float16*)(Q + 128);
  const _Float16* S3h  = Hq;
  const _Float16* V3h  = Hq + 8;
  const _Float16* D13h = Hq + 32;
  const _Float16* D23h = Hq + 96;

  const v8h S3tab = *(const v8h*)S3h;

  // ---- Scalar GEMM: out[:, 0:8] = F_s[N,2048] @ W4s * 1/sqrt(2048) --------
  {
    v8f acc = {};
#pragma unroll
    for (int b = 0; b < 4; ++b) {
      for (int kcl = 0; kcl < 16; ++kcl) {          // 16 chunks of K=32 per block
        const int kc = b * 16 + kcl;
        v16h afrag;
#pragma unroll
        for (int part = 0; part < 2; ++part) {
          const int idx = kc * 32 + part * 16 + h * 8;   // 8-aligned run
          const int u = (idx >> 6) & 7;
          const int v = (idx >> 3) & 7;
          v8h f;
          if (b == 0) {                 // (n1s x n2s) x n3s
            _Float16 p = (_Float16)(S1[u] * S2[v]);
            f = S3tab * p;
          } else if (b == 1) {          // (n1s x n2v) . n3v / sqrt3
            _Float16 p = (_Float16)(S1[u] * INV_SQRT3);
            f = (*(const v8h*)(D23h + v * 8)) * p;
          } else if (b == 2) {          // (n1v x n2s) . n3v / sqrt3
            _Float16 p = (_Float16)(S2[v] * INV_SQRT3);
            f = (*(const v8h*)(D13h + u * 8)) * p;
          } else {                      // (n1v.n2v)/sqrt3 x n3s
            _Float16 p = (_Float16)(D12[u * 8 + v] * INV_SQRT3);
            f = S3tab * p;
          }
#pragma unroll
          for (int j = 0; j < 8; ++j) afrag[part * 8 + j] = f[j];
        }
        const v16h bfrag = *(const v16h*)(frag_s + (size_t)(kc * 32 + lane) * 16);
        __builtin_prefetch(frag_s + (size_t)((kc + 2) * 32 + lane) * 16, 0, 1);
        acc = __builtin_amdgcn_wmma_f32_16x16x32_f16(false, afrag, false, bfrag,
                                                     (short)0, acc, false, false);
      }
    }
    if (ncol < 8) {
#pragma unroll
      for (int j = 0; j < 8; ++j) {
        const int M = h ? (j + 8) : j;
        out[(size_t)(obase + M) * 32 + ncol] = acc[j] * SSCALE;
      }
    }
  }

  // ---- Vector GEMMs (k=0..2): out[:, 8+w*3+k] = F_v[:,:,k] @ W4v ----------
#pragma unroll 1
  for (int k = 0; k < 3; ++k) {
    const v8h V3tab = *(const v8h*)(V3h + 8 * k);
    v8f acc = {};
#pragma unroll
    for (int b = 0; b < 6; ++b) {
      for (int kcl = 0; kcl < 16; ++kcl) {
        const int kc = b * 16 + kcl;
        v16h afrag;
#pragma unroll
        for (int part = 0; part < 2; ++part) {
          const int idx = kc * 32 + part * 16 + h * 8;
          const int u = (idx >> 6) & 7;
          const int v = (idx >> 3) & 7;
          v8h f;
          if (b == 0) {                 // (n1s x n2s) x n3v
            _Float16 p = (_Float16)(S1[u] * S2[v]);
            f = V3tab * p;
          } else if (b == 1) {          // (n1s x n2v) x n3s
            _Float16 p = (_Float16)(S1[u] * V2[v * 3 + k]);
            f = S3tab * p;
          } else if (b == 2) {          // (n1v x n2s) x n3s
            _Float16 p = (_Float16)(V1[u * 3 + k] * S2[v]);
            f = S3tab * p;
          } else if (b == 3) {          // (n1v.n2v)/sqrt3 x n3v
            _Float16 p = (_Float16)(D12[u * 8 + v] * INV_SQRT3);
            f = V3tab * p;
          } else if (b == 4) {          // ((n1v x n2v) x n3v)/2 via dot tables
            _Float16 p1 = (_Float16)(0.5f * V2[v * 3 + k]);
            _Float16 p2 = (_Float16)(-0.5f * V1[u * 3 + k]);
            f = (*(const v8h*)(D13h + u * 8)) * p1
              + (*(const v8h*)(D23h + v * 8)) * p2;
          } else {                      // 2e path via symmetric-traceless projector
            _Float16 p1 = (_Float16)(C5A * V2[v * 3 + k]);
            _Float16 p2 = (_Float16)(C5A * V1[u * 3 + k]);
            _Float16 p3 = (_Float16)(-C5B * D12[u * 8 + v]);
            f = (*(const v8h*)(D13h + u * 8)) * p1
              + (*(const v8h*)(D23h + v * 8)) * p2
              + V3tab * p3;
          }
#pragma unroll
          for (int j = 0; j < 8; ++j) afrag[part * 8 + j] = f[j];
        }
        const v16h bfrag = *(const v16h*)(frag_v + (size_t)(kc * 32 + lane) * 16);
        __builtin_prefetch(frag_v + (size_t)((kc + 2) * 32 + lane) * 16, 0, 1);
        acc = __builtin_amdgcn_wmma_f32_16x16x32_f16(false, afrag, false, bfrag,
                                                     (short)0, acc, false, false);
      }
    }
    if (ncol < 8) {
#pragma unroll
      for (int j = 0; j < 8; ++j) {
        const int M = h ? (j + 8) : j;
        out[(size_t)(obase + M) * 32 + 8 + ncol * 3 + k] = acc[j] * VSCALE;
      }
    }
  }
}

extern "C" void kernel_launch(void* const* d_in, const int* in_sizes, int n_in,
                              void* d_out, int out_size, void* d_ws, size_t ws_size,
                              hipStream_t stream) {
  const float* x   = (const float*)d_in[0];
  const float* W1s = (const float*)d_in[1];
  const float* W1v = (const float*)d_in[2];
  const float* W2s = (const float*)d_in[3];
  const float* W2v = (const float*)d_in[4];
  const float* W3s = (const float*)d_in[5];
  const float* W3v = (const float*)d_in[6];
  const float* W4s = (const float*)d_in[7];
  const float* W4v = (const float*)d_in[8];
  float* out = (float*)d_out;

  _Float16* frag_s = (_Float16*)d_ws;                 // 64*32*16 halves = 64 KB
  _Float16* frag_v = frag_s + (size_t)64 * 32 * 16;   // 96*32*16 halves = 96 KB

  // Fold W4 into f16 B-fragment layout (runs every launch; deterministic).
  fold_w4_kernel<<<20, 256, 0, stream>>>(W4s, W4v, frag_s, frag_v);

  // Fused tensor-product + WMMA GEMM: 64 blocks x 256 threads, 128 rows each.
  const size_t shmem = (size_t)SHMEM_FLOATS * sizeof(float);  // ~107.5 KB < 320 KB/WGP
  tp3body_kernel<<<8192 / ROWS_PER_BLOCK, 256, shmem, stream>>>(
      x, W1s, W1v, W2s, W2v, W3s, W3v, frag_s, frag_v, out);
}
